// NodeAttention_16269336117301
// MI455X (gfx1250) — matrix operand order repaired and checked by reading, not compiled, not run on hardware
//
#include <hip/hip_runtime.h>
#include <hip/hip_bf16.h>
#include <math.h>

typedef __attribute__((ext_vector_type(16))) _Float16 v16h;
typedef __attribute__((ext_vector_type(8)))  float    v8f;

#define NNODES 20000
#define KZ  16
#define IFZ 256
#define AHZ 8
#define AFZ 32
#define HF  256
#define LN_EPS 1e-5f
#define NB  8     // nodes per workgroup
#define FS  24    // fragment lane stride in halfs (16 data + 8 pad -> 48B, <=2-way LDS conflict)

union FragA { v16h v; uint4 q[2]; };
union H4   { _Float16 h[4]; uint2 u; };
union H8   { _Float16 h[8]; uint4 u; };

// ---------------------------------------------------------------------------
// Kernel 1: convert the 5 big 256x256 f32 weight matrices to f16 in workspace
// ---------------------------------------------------------------------------
__global__ void na_convert_weights(const float* __restrict__ Wq,
                                   const float* __restrict__ Wk,
                                   const float* __restrict__ Wv,
                                   const float* __restrict__ Wg,
                                   const float* __restrict__ Wback,
                                   _Float16* __restrict__ dst) {
  int which = blockIdx.y;
  int idx = blockIdx.x * blockDim.x + threadIdx.x;   // 0..65535
  const float* src = (which == 0) ? Wq : (which == 1) ? Wk :
                     (which == 2) ? Wv : (which == 3) ? Wg : Wback;
  dst[(size_t)which * IFZ * HF + idx] = (_Float16)src[idx];
}

// ---------------------------------------------------------------------------
// Kernel 2: bias2[row][h] = (layernorm(x_2[row]) @ Wb)[h]    row in [0, N*KZ)
// one wave32 per row; streams x_2 exactly once
// ---------------------------------------------------------------------------
__global__ __launch_bounds__(256)
void na_bias_kernel(const float* __restrict__ x2,
                    const float* __restrict__ Wb,
                    const float* __restrict__ blng,
                    const float* __restrict__ blnb,
                    float* __restrict__ bias2) {
  int lane = threadIdx.x & 31;
  int wv   = threadIdx.x >> 5;
  size_t row = (size_t)blockIdx.x * 8 + wv;          // < 320000

  const float* xr = x2 + row * IFZ + lane * 8;
  float4 a0 = *(const float4*)xr;
  float4 a1 = *(const float4*)(xr + 4);
  float x[8] = {a0.x, a0.y, a0.z, a0.w, a1.x, a1.y, a1.z, a1.w};

  float s1 = 0.f, s2 = 0.f;
  for (int i = 0; i < 8; i++) { s1 += x[i]; s2 += x[i] * x[i]; }
  for (int off = 16; off >= 1; off >>= 1) {
    s1 += __shfl_xor(s1, off, 32);
    s2 += __shfl_xor(s2, off, 32);
  }
  float mean = s1 * (1.f / IFZ);
  float rstd = rsqrtf(s2 * (1.f / IFZ) - mean * mean + LN_EPS);

  float acc[8] = {0.f, 0.f, 0.f, 0.f, 0.f, 0.f, 0.f, 0.f};
  for (int i = 0; i < 8; i++) {
    int e = lane * 8 + i;
    float xn = (x[i] - mean) * rstd * blng[e] + blnb[e];
    const float4* wr = (const float4*)(Wb + (size_t)e * AHZ);
    float4 w0 = wr[0], w1 = wr[1];
    acc[0] += xn * w0.x; acc[1] += xn * w0.y; acc[2] += xn * w0.z; acc[3] += xn * w0.w;
    acc[4] += xn * w1.x; acc[5] += xn * w1.y; acc[6] += xn * w1.z; acc[7] += xn * w1.w;
  }
  for (int h = 0; h < 8; h++)
    for (int off = 16; off >= 1; off >>= 1)
      acc[h] += __shfl_xor(acc[h], off, 32);

  if (lane == 0) {
    float* o = bias2 + row * AHZ;
    float4 o0 = {acc[0], acc[1], acc[2], acc[3]};
    float4 o1 = {acc[4], acc[5], acc[6], acc[7]};
    *(float4*)o       = o0;
    *(float4*)(o + 4) = o1;
  }
}

// ---------------------------------------------------------------------------
// Kernel 3: fused gather + RoPE attention + gate + Wback + residual + LN
// One workgroup (8 waves / 256 threads) handles NB=8 nodes; wave w owns node w.
// K/V projections via v_wmma_f32_16x16x32_f16: per (node,head) a [16,256]@[256,32]
// GEMM = 8 K-steps x (2 K-tiles + 2 V-tiles) = 32 WMMAs.
// ---------------------------------------------------------------------------
__global__ __launch_bounds__(256)
void na_attn_kernel(const float* __restrict__ x1,
                    const float* __restrict__ pos_emb,
                    const int*   __restrict__ eidx,
                    const _Float16* __restrict__ Wq16,
                    const _Float16* __restrict__ Wk16,
                    const _Float16* __restrict__ Wv16,
                    const _Float16* __restrict__ Wg16,
                    const _Float16* __restrict__ Wback16,
                    const float* __restrict__ bg,
                    const float* __restrict__ bback,
                    const float* __restrict__ ln1g,
                    const float* __restrict__ ln1b,
                    const float* __restrict__ bias2,
                    float* __restrict__ out) {
  extern __shared__ char smem[];
  _Float16* Af  = (_Float16*)smem;                   // NB*8*32*FS halfs (A frags, gathered neighbors)
  _Float16* Bkf = Af  + NB * 8 * 32 * FS;            // 2*8*32*FS halfs (Wk head-block frags)
  _Float16* Bvf = Bkf + 2 * 8 * 32 * FS;             // 2*8*32*FS halfs (Wv head-block frags)
  float* cosn = (float*)(Bvf + 2 * 8 * 32 * FS);     // NB*KZ*AFZ
  float* sinn = cosn + NB * KZ * AFZ;                // NB*KZ*AFZ
  float* x1s  = sinn + NB * KZ * AFZ;                // NB*IFZ
  float* qs   = x1s + NB * IFZ;                      // NB*IFZ (q, later reused as pre-LN tmp)
  float* gts  = qs  + NB * IFZ;                      // NB*IFZ (gate)
  float* outs = gts + NB * IFZ;                      // NB*IFZ (gated attention output)

  const int tid   = threadIdx.x;
  const int lane  = tid & 31;
  const int wv    = tid >> 5;
  const int node0 = blockIdx.x * NB;

  // --- 1. stage x1 rows of our 8 nodes (f32, for q/gate/residual) ------------
  for (int i = 0; i < NB; i++) {
    int e = tid + i * 256;
    x1s[e] = x1[(size_t)node0 * IFZ + e];
  }

  // --- 2. gather neighbor rows -> f16 A-matrix fragments ---------------------
  // A layout (16-bit 16x32): lane = row + 16*((k>>3)&1), half j = (k&7) + 8*(k>>4)
  {
    int row  = tid >> 1;              // 0..127 -> (node, kz)
    int half = tid & 1;
    int nd = row >> 4, r = row & 15;
    int src = eidx[(node0 + nd) * KZ + r];
    const float* xr = x1 + (size_t)src * IFZ + half * 128;
    for (int kk = 0; kk < 32; kk++) {
      int k = half * 128 + kk * 4;
      float4 f = *(const float4*)(xr + kk * 4);
      int ks = k >> 5, kr = k & 31;
      int laned = r + 16 * ((kr >> 3) & 1);
      int j = (kr & 7) + 8 * (kr >> 4);
      H4 t;
      t.h[0] = (_Float16)f.x; t.h[1] = (_Float16)f.y;
      t.h[2] = (_Float16)f.z; t.h[3] = (_Float16)f.w;
      *(uint2*)&Af[((nd * 8 + ks) * 32 + laned) * FS + j] = t.u;
    }
  }

  // --- 3. gathered neighbor pos_emb -> cos/sin tables (shared across heads) --
  for (int i = 0; i < 16; i++) {
    int e = tid + i * 256;            // < NB*KZ*AFZ = 4096
    int nd = e >> 9, rem = e & 511, r = rem >> 5, f = rem & 31;
    int src = eidx[(node0 + nd) * KZ + r];
    float p = pos_emb[(size_t)src * AFZ + f];
    cosn[e] = cosf(p);
    sinn[e] = sinf(p);
  }
  __syncthreads();

  // --- 4. q and gate projections (vector f32, small GEMMs) -------------------
  {
    int nd = tid >> 5;
    int c0 = (tid & 31) * 8;
    float aq[8] = {0.f, 0.f, 0.f, 0.f, 0.f, 0.f, 0.f, 0.f};
    float ag[8] = {0.f, 0.f, 0.f, 0.f, 0.f, 0.f, 0.f, 0.f};
    for (int k = 0; k < IFZ; k++) {
      float a = x1s[nd * IFZ + k];
      H8 wq, wg;
      wq.u = *(const uint4*)(Wq16 + (size_t)k * HF + c0);
      wg.u = *(const uint4*)(Wg16 + (size_t)k * HF + c0);
      for (int j = 0; j < 8; j++) {
        aq[j] += a * (float)wq.h[j];
        ag[j] += a * (float)wg.h[j];
      }
    }
    for (int j = 0; j < 8; j++) {
      qs[nd * IFZ + c0 + j] = aq[j];
      float gv = ag[j] + bg[c0 + j];
      gts[nd * IFZ + c0 + j] = 1.f / (1.f + expf(-gv));
    }
  }
  __syncthreads();

  // --- 5. RoPE on q (pairs f, f+16 inside each 32-wide head) -----------------
  for (int i = 0; i < 4; i++) {
    int e = tid + i * 256;            // < NB*AHZ*16 = 1024
    int nd = e >> 7, rem = e & 127, h = rem >> 4, f = rem & 15;
    int c = h * AFZ + f;
    float p0 = pos_emb[(size_t)(node0 + nd) * AFZ + f];
    float p1 = pos_emb[(size_t)(node0 + nd) * AFZ + f + 16];
    float q0 = qs[nd * IFZ + c], q1 = qs[nd * IFZ + c + 16];
    qs[nd * IFZ + c]      = q0 * cosf(p0) - q1 * sinf(p0);
    qs[nd * IFZ + c + 16] = q1 * cosf(p1) + q0 * sinf(p1);
  }

  const int mynode = wv;
  const int hi = lane >> 4;           // which 8-row half this lane owns in C/D
  const int c  = lane & 15;           // column within 16-wide D tile
  const float invs = 0.17677669529663687f;   // 1/sqrt(AFZ)

  // --- 6. per-head loop: WMMA K/V projection + RoPE + softmax + weighted V ---
  for (int h = 0; h < AHZ; h++) {
    __syncthreads();                  // previous head's frag reads complete
    // stage Wk/Wv head column-block as B fragments:
    // B layout (16-bit 32x16): lane = col + 16*((k>>3)&1), half j = (k&7) + 8*(k>>4)
    for (int i = 0; i < 32; i++) {
      int idx = tid + i * 256;        // < 8192
      int k = idx >> 5, cc = idx & 31;
      int sub = cc >> 4, col = cc & 15, ks = k >> 5, kr = k & 31;
      int doff = ((sub * 8 + ks) * 32 + (col + ((kr >> 3) & 1) * 16)) * FS + ((kr & 7) + 8 * (kr >> 4));
      Bkf[doff] = Wk16[(size_t)k * HF + h * AFZ + cc];
      Bvf[doff] = Wv16[(size_t)k * HF + h * AFZ + cc];
    }
    __syncthreads();

    v8f k0 = {0.f, 0.f, 0.f, 0.f, 0.f, 0.f, 0.f, 0.f};
    v8f k1 = k0, v0 = k0, v1 = k0;
    for (int ks = 0; ks < 8; ks++) {
      FragA a, bk0, bk1, bv0, bv1;
      const uint4* ap = (const uint4*)&Af[((mynode * 8 + ks) * 32 + lane) * FS];
      a.q[0] = ap[0]; a.q[1] = ap[1];
      const uint4* p0 = (const uint4*)&Bkf[(ks * 32 + lane) * FS];
      bk0.q[0] = p0[0]; bk0.q[1] = p0[1];
      const uint4* p1 = (const uint4*)&Bkf[(((8 + ks)) * 32 + lane) * FS];
      bk1.q[0] = p1[0]; bk1.q[1] = p1[1];
      const uint4* p2 = (const uint4*)&Bvf[(ks * 32 + lane) * FS];
      bv0.q[0] = p2[0]; bv0.q[1] = p2[1];
      const uint4* p3 = (const uint4*)&Bvf[(((8 + ks)) * 32 + lane) * FS];
      bv1.q[0] = p3[0]; bv1.q[1] = p3[1];

      k0 = __builtin_amdgcn_wmma_f32_16x16x32_f16(false, a.v, false, bk0.v, (short)0, k0, false, false);
      k1 = __builtin_amdgcn_wmma_f32_16x16x32_f16(false, a.v, false, bk1.v, (short)0, k1, false, false);
      v0 = __builtin_amdgcn_wmma_f32_16x16x32_f16(false, a.v, false, bv0.v, (short)0, v0, false, false);
      v1 = __builtin_amdgcn_wmma_f32_16x16x32_f16(false, a.v, false, bv1.v, (short)0, v1, false, false);
    }

    // RoPE k + scores.  D layout: row m = r + 8*hi, col = c (tile0) / c+16 (tile1)
    float qv0 = qs[mynode * IFZ + h * AFZ + c];
    float qv1 = qs[mynode * IFZ + h * AFZ + c + 16];
    float p[8];
    for (int r = 0; r < 8; r++) {
      int m = r + 8 * hi;
      int base = (mynode * KZ + m) * AFZ;
      float c0v = cosn[base + c],      s0v = sinn[base + c];
      float c1v = cosn[base + c + 16], s1v = sinn[base + c + 16];
      float kn0 = k0[r] * c0v - k1[r] * s0v;
      float kn1 = k1[r] * c1v + k0[r] * s1v;
      p[r] = kn0 * qv0 + kn1 * qv1;
    }
    // reduce the 32-wide head dot across the 16 lanes of each half-wave
    for (int off = 8; off >= 1; off >>= 1)
      for (int r = 0; r < 8; r++)
        p[r] += __shfl_xor(p[r], off, 32);

    float s[8], mx = -1e30f;
    for (int r = 0; r < 8; r++) {
      int m = r + 8 * hi;
      s[r] = p[r] * invs + bias2[((size_t)(node0 + mynode) * KZ + m) * AHZ + h];
      mx = fmaxf(mx, s[r]);
    }
    mx = fmaxf(mx, __shfl_xor(mx, 16, 32));     // combine rows 0-7 with 8-15
    float w[8], se = 0.f;
    for (int r = 0; r < 8; r++) { w[r] = expf(s[r] - mx); se += w[r]; }
    se += __shfl_xor(se, 16, 32);
    float rs = 1.f / se;

    float po0 = 0.f, po1 = 0.f;
    for (int r = 0; r < 8; r++) {
      po0 += w[r] * rs * v0[r];
      po1 += w[r] * rs * v1[r];
    }
    po0 += __shfl_xor(po0, 16, 32);             // rows 0-7 + rows 8-15
    po1 += __shfl_xor(po1, 16, 32);
    if (hi == 0) {
      int oc = mynode * IFZ + h * AFZ + c;
      outs[oc]      = gts[oc]      * po0;
      outs[oc + 16] = gts[oc + 16] * po1;
    }
  }
  __syncthreads();

  // --- 7. out @ Wback + bback + sqrt(2)*x1 (into qs, reused as tmp) ----------
  {
    int nd = tid >> 5;
    int c0 = (tid & 31) * 8;
    float acc[8] = {0.f, 0.f, 0.f, 0.f, 0.f, 0.f, 0.f, 0.f};
    for (int k = 0; k < IFZ; k++) {
      float a = outs[nd * IFZ + k];
      H8 wb;
      wb.u = *(const uint4*)(Wback16 + (size_t)k * HF + c0);
      for (int j = 0; j < 8; j++) acc[j] += a * (float)wb.h[j];
    }
    for (int j = 0; j < 8; j++) {
      int cc = c0 + j;
      qs[nd * IFZ + cc] = acc[j] + bback[cc] + 1.4142135623730951f * x1s[nd * IFZ + cc];
    }
  }
  __syncthreads();

  // --- 8. final layernorm, one wave per node ---------------------------------
  {
    int nd = wv;
    float xv[8], s1 = 0.f, s2 = 0.f;
    for (int i = 0; i < 8; i++) {
      xv[i] = qs[nd * IFZ + lane * 8 + i];
      s1 += xv[i]; s2 += xv[i] * xv[i];
    }
    for (int off = 16; off >= 1; off >>= 1) {
      s1 += __shfl_xor(s1, off, 32);
      s2 += __shfl_xor(s2, off, 32);
    }
    float mean = s1 * (1.f / IFZ);
    float rstd = rsqrtf(s2 * (1.f / IFZ) - mean * mean + LN_EPS);
    for (int i = 0; i < 8; i++) {
      int e = lane * 8 + i;
      out[(size_t)(node0 + nd) * IFZ + e] = (xv[i] - mean) * rstd * ln1g[e] + ln1b[e];
    }
  }
}

// ---------------------------------------------------------------------------
extern "C" void kernel_launch(void* const* d_in, const int* in_sizes, int n_in,
                              void* d_out, int out_size, void* d_ws, size_t ws_size,
                              hipStream_t stream) {
  const float* x1    = (const float*)d_in[0];
  const float* x2    = (const float*)d_in[1];
  const float* pe    = (const float*)d_in[2];
  const int*   ei    = (const int*)d_in[3];
  const float* Wq    = (const float*)d_in[4];
  const float* Wk    = (const float*)d_in[5];
  const float* Wv    = (const float*)d_in[6];
  const float* Wb    = (const float*)d_in[7];
  const float* blng  = (const float*)d_in[8];
  const float* blnb  = (const float*)d_in[9];
  const float* Wg    = (const float*)d_in[10];
  const float* bg    = (const float*)d_in[11];
  const float* Wback = (const float*)d_in[12];
  const float* bback = (const float*)d_in[13];
  const float* ln1g  = (const float*)d_in[14];
  const float* ln1b  = (const float*)d_in[15];

  char* ws = (char*)d_ws;
  _Float16* w16 = (_Float16*)ws;                         // 5 x 65536 halfs
  float* bias2  = (float*)(ws + (size_t)5 * IFZ * HF * 2);

  na_convert_weights<<<dim3(256, 5), 256, 0, stream>>>(Wq, Wk, Wv, Wg, Wback, w16);
  na_bias_kernel<<<dim3((NNODES * KZ) / 8), 256, 0, stream>>>(x2, Wb, blng, blnb, bias2);

  size_t smem = (size_t)(NB * 8 * 32 * FS + 2 * (2 * 8 * 32 * FS)) * 2   // A + Bk + Bv frags
              + (size_t)(2 * NB * KZ * AFZ + 4 * NB * IFZ) * 4;          // cos/sin + x1/q/gate/out
  na_attn_kernel<<<dim3(NNODES / NB), 256, smem, stream>>>(
      x1, pe, ei,
      w16 + 0 * 65536, w16 + 1 * 65536, w16 + 2 * 65536,
      w16 + 3 * 65536, w16 + 4 * 65536,
      bg, bback, ln1g, ln1b, bias2, (float*)d_out);
}
